// MambaBlock_25898652795143
// MI455X (gfx1250) — matrix operand order repaired and checked by reading, not compiled, not run on hardware
//
#include <hip/hip_runtime.h>
#include <hip/hip_bf16.h>
#include <stdint.h>

// ---------------- Types for WMMA / TDM ----------------
typedef __bf16 bf16x16 __attribute__((ext_vector_type(16)));
typedef float  f32x8   __attribute__((ext_vector_type(8)));
typedef unsigned int u32x4 __attribute__((ext_vector_type(4)));
typedef int          i32x8 __attribute__((ext_vector_type(8)));
typedef int          i32x4 __attribute__((ext_vector_type(4)));

union FragBF {            // 32 bytes: one WMMA A/B operand (16 bf16 per lane)
    bf16x16 v;
    uint4   q[2];
    unsigned short u[16];
};

__device__ __forceinline__ unsigned short f2bf(float f) {
    unsigned int u = __float_as_uint(f);
    unsigned int r = (u + 0x7FFFu + ((u >> 16) & 1u)) >> 16;   // RNE
    return (unsigned short)r;
}
__device__ __forceinline__ float silu(float v) {
    return v / (1.f + __expf(-v));
}

// ---- CDNA5 async global->LDS copy (16B per lane), tracked by ASYNCcnt ----
__device__ __forceinline__ void async_b128(void* lds, const void* g) {
    unsigned           loff = (unsigned)(uintptr_t)lds;   // addr[31:0] = LDS offset
    unsigned long long ga   = (unsigned long long)(uintptr_t)g;
    asm volatile("global_load_async_to_lds_b128 %0, %1, off"
                 :: "v"(loff), "v"(ga) : "memory");
}
template <int N>
__device__ __forceinline__ void wait_asynccnt() {
    asm volatile("s_wait_asynccnt %0" :: "i"(N) : "memory");
}
template <int N>
__device__ __forceinline__ void wait_tensorcnt() {
    __builtin_amdgcn_s_wait_tensorcnt((short)N);
}

// ---- CDNA5 Tensor Data Mover: DMA a 2D tile (tile_d0 contiguous elems x
// tile_d1 rows, row stride stride0 elems) from global memory into LDS.
// D# bit layout per ISA cdna5 ch.8: group0 {count,lds_addr,global_addr,type=2},
// group1 {data_size, tensor dims (= tile dims, all in-bounds), tile dims, stride}.
// dsize_code: 0=1B 1=2B 2=4B 3=8B.  Tracked by TENSORcnt; EXEC ignored.
// This toolchain exposes the 6-arg builtin: (g0, g1, g2, g3, g4, cpol).
__device__ __forceinline__ void tdm_load_2d(unsigned lds_addr, const void* g,
                                            unsigned tile_d0, unsigned tile_d1,
                                            unsigned long long stride0,
                                            unsigned dsize_code) {
    unsigned long long ga = (unsigned long long)(uintptr_t)g;
    u32x4 g0;
    g0.x = 1u;                                            // count=1, user mode
    g0.y = lds_addr;                                      // lds_addr (bytes)
    g0.z = (unsigned)ga;                                  // global_addr[31:0]
    g0.w = ((unsigned)(ga >> 32) & 0x01FFFFFFu) | (2u << 30);  // addr[56:32] | type=2
    i32x8 g1;
    g1[0] = (int)(dsize_code << 16);                      // wg_mask=0 | data_size
    g1[1] = (int)((tile_d0 & 0xFFFFu) << 16);             // abar=0 | tensor_dim0 lo
    g1[2] = (int)((tile_d0 >> 16) | ((tile_d1 & 0xFFFFu) << 16)); // t_d0 hi | t_d1 lo
    g1[3] = (int)((tile_d1 >> 16) | ((tile_d0 & 0xFFFFu) << 16)); // t_d1 hi | tile_dim0
    g1[4] = (int)(tile_d1 & 0xFFFFu);                     // tile_dim1 | tile_dim2=0
    g1[5] = (int)(unsigned)stride0;                       // tensor_dim0_stride lo
    g1[6] = (int)((stride0 >> 32) & 0xFFFFu);             // stride hi | dim1_stride=0
    g1[7] = 0;
    i32x4 z4 = {0, 0, 0, 0};
    i32x8 z8 = {0, 0, 0, 0, 0, 0, 0, 0};
    __builtin_amdgcn_tensor_load_to_lds(g0, g1, z4, z4, z8, 0);
}

// ---------------- fp32 -> bf16 conversion ----------------
__global__ void k_f32_to_bf16(const float* __restrict__ in,
                              unsigned short* __restrict__ out, size_t n) {
    size_t i = (size_t)blockIdx.x * 256 + threadIdx.x;
    if (i < n) out[i] = f2bf(in[i]);
}

// transpose-convert:  in (K x N) fp32  ->  out (Npad x K) bf16, zero-padded rows
__global__ void k_f32_to_bf16_T(const float* __restrict__ in,
                                unsigned short* __restrict__ out,
                                int K, int N, int Npad) {
    size_t i = (size_t)blockIdx.x * 256 + threadIdx.x;
    size_t total = (size_t)Npad * K;
    if (i >= total) return;
    int n = (int)(i / K), k = (int)(i % K);
    out[i] = (n < N) ? f2bf(in[(size_t)k * N + n]) : (unsigned short)0;
}

// ---------------- LayerNorm (D=1024, 256 threads/row, wave32 reductions) ----
__global__ void k_layernorm_bf16(const float* __restrict__ x,
                                 const float* __restrict__ gamma,
                                 const float* __restrict__ beta,
                                 unsigned short* __restrict__ out) {
    const int row = blockIdx.x;
    const int tid = threadIdx.x;
    const float* xr = x + (size_t)row * 1024;

    float v[4], s = 0.f, sq = 0.f;
#pragma unroll
    for (int i = 0; i < 4; ++i) {
        v[i] = xr[tid + i * 256];
        s += v[i];
        sq += v[i] * v[i];
    }
#pragma unroll
    for (int off = 16; off > 0; off >>= 1) {        // wave32 reduce
        s  += __shfl_xor(s,  off, 32);
        sq += __shfl_xor(sq, off, 32);
    }
    __shared__ float ls[8], lq[8];
    const int wave = tid >> 5, lane = tid & 31;
    if (lane == 0) { ls[wave] = s; lq[wave] = sq; }
    __syncthreads();
    if (wave == 0) {
        float a  = (lane < 8) ? ls[lane] : 0.f;
        float b2 = (lane < 8) ? lq[lane] : 0.f;
#pragma unroll
        for (int off = 4; off > 0; off >>= 1) {
            a  += __shfl_xor(a,  off, 32);
            b2 += __shfl_xor(b2, off, 32);
        }
        if (lane == 0) { ls[0] = a; lq[0] = b2; }
    }
    __syncthreads();
    const float mean = ls[0] * (1.f / 1024.f);
    const float var  = lq[0] * (1.f / 1024.f) - mean * mean;
    const float rstd = rsqrtf(var + 1e-5f);
#pragma unroll
    for (int i = 0; i < 4; ++i) {
        int c = tid + i * 256;
        out[(size_t)row * 1024 + c] = f2bf((v[i] - mean) * rstd * gamma[c] + beta[c]);
    }
}

// ---------------- bf16 WMMA GEMM: C(MxN) = A(MxK) * Bt(NxK)^T ----------------
// 128x128 block tile, K-step 32, 8 waves, each wave = 64x32 (4x2 WMMA tiles).
// LDS tiles are DMA'd by the Tensor Data Mover (one descriptor per 8KB tile),
// double-buffered; wave 0 issues/waits TENSORcnt, barrier releases the block.
// Steady-state loop is branch-free (last iteration peeled).
// EPI: 0 = plain f32 store; 1 = softplus(c + bias[col]); 2 = c + resid[idx].
template <int EPI>
__global__ __launch_bounds__(256)
void k_gemm_bf16(const unsigned short* __restrict__ A,
                 const unsigned short* __restrict__ Bt,   // N x K, K contiguous
                 float* __restrict__ C,
                 int M, int N, int K, int lda, int ldc,
                 const float* __restrict__ bias,
                 const float* __restrict__ resid) {
    __shared__ __align__(16) unsigned short lsA[2][128 * 32];   // [m][k]
    __shared__ __align__(16) unsigned short lsB[2][128 * 32];   // [n][k]

    const int tid  = threadIdx.x;
    const int lane = tid & 31;
    const int wave = tid >> 5;
    const int wm   = wave >> 2;        // 0..1  -> 64-row strip
    const int wn   = wave & 3;         // 0..3  -> 32-col strip
    const int half = lane >> 4;        // hi/lo half-wave
    const int r16  = lane & 15;

    const int m0 = blockIdx.y * 128;
    const int n0 = blockIdx.x * 128;

    auto issue_tile = [&](int k0, int buf) {   // wave 0 only: 2 TDM descriptors
        tdm_load_2d((unsigned)(uintptr_t)&lsA[buf][0],
                    A + (size_t)m0 * lda + k0, 32, 128, (unsigned long long)lda, 1);
        tdm_load_2d((unsigned)(uintptr_t)&lsB[buf][0],
                    Bt + (size_t)n0 * K + k0, 32, 128, (unsigned long long)K, 1);
    };

    f32x8 acc[4][2] = {};
    const int KT = K / 32;

    auto compute = [&](int buf) {
        FragBF af[4], bfg[2];
#pragma unroll
        for (int tm = 0; tm < 4; ++tm) {
            int row = wm * 64 + tm * 16 + r16;
            // 16-bit A 16x32 layout: u[0..7]=K[8h..8h+7], u[8..15]=K[16+8h..+7]
            af[tm].q[0] = *(const uint4*)&lsA[buf][row * 32 + 8 * half];
            af[tm].q[1] = *(const uint4*)&lsA[buf][row * 32 + 16 + 8 * half];
        }
#pragma unroll
        for (int tn = 0; tn < 2; ++tn) {
            int col = wn * 32 + tn * 16 + r16;
            // B 32x16 layout: u[0..7]=K[8h..8h+7], u[8..15]=K[16+8h..+7]
            bfg[tn].q[0] = *(const uint4*)&lsB[buf][col * 32 + 8 * half];
            bfg[tn].q[1] = *(const uint4*)&lsB[buf][col * 32 + 16 + 8 * half];
        }
#pragma unroll
        for (int tm = 0; tm < 4; ++tm)
#pragma unroll
            for (int tn = 0; tn < 2; ++tn)
                acc[tm][tn] = __builtin_amdgcn_wmma_f32_16x16x32_bf16(
                    false, af[tm].v, false, bfg[tn].v,
                    (short)0, acc[tm][tn], false, false);
    };

    if (wave == 0) issue_tile(0, 0);
    int kt = 0;
    for (; kt + 1 < KT; ++kt) {                 // branch-free steady state
        const int buf = kt & 1;
        if (wave == 0) {
            issue_tile((kt + 1) * 32, buf ^ 1); // prefetch next tile pair
            wait_tensorcnt<2>();                // current pair (in-order) done
        }
        __syncthreads();
        compute(buf);
        __syncthreads();                        // protect buf^1 before next issue
    }
    if (wave == 0) wait_tensorcnt<0>();         // peeled last tile
    __syncthreads();
    compute(kt & 1);

    // ---- epilogue + store (C layout: VGPR e -> row e+8*half, col lane&15) ----
#pragma unroll
    for (int tm = 0; tm < 4; ++tm) {
#pragma unroll
        for (int tn = 0; tn < 2; ++tn) {
            int colg = n0 + wn * 32 + tn * 16 + r16;
#pragma unroll
            for (int e = 0; e < 8; ++e) {
                int rowg = m0 + wm * 64 + tm * 16 + 8 * half + e;
                size_t idx = (size_t)rowg * ldc + colg;
                float v = acc[tm][tn][e];
                if (EPI == 1) {
                    v += bias[colg];
                    v = (v > 20.f) ? v : log1pf(__expf(v));
                } else if (EPI == 2) {
                    v += resid[idx];
                }
                C[idx] = v;
            }
        }
    }
}

// ---------------- depthwise causal conv (K=4) + SiLU -------------------------
// u lives in xz[:, 0:2048] (row stride 4096); writes fp32 + bf16 copies.
__global__ void k_conv_silu(const float* __restrict__ xz,
                            const float* __restrict__ cw,
                            const float* __restrict__ cb,
                            float* __restrict__ u32,
                            unsigned short* __restrict__ ubf) {
    size_t idx = (size_t)blockIdx.x * 256 + threadIdx.x;
    const size_t total = (size_t)8 * 2048 * 2048;
    if (idx >= total) return;
    int c = (int)(idx % 2048);
    size_t bl = idx / 2048;                 // b*2048 + l
    int l = (int)(bl % 2048);

    float acc = cb[c];
#pragma unroll
    for (int k = 0; k < 4; ++k) {
        int li = l - 3 + k;
        if (li >= 0)
            acc += xz[(bl - (size_t)(3 - k)) * 4096 + c] * cw[c * 4 + k];
    }
    float s = silu(acc);
    u32[idx] = s;
    ubf[idx] = f2bf(s);
}

// ---------------- selective scan, fused with gating ---------------------------
// One thread per (b, d) channel; h[16] in VGPRs. Bm/Cm chunks (64 steps) are
// prefetched into double-buffered LDS via GLOBAL_LOAD_ASYNC_TO_LDS (ASYNCcnt),
// one chunk ahead of the sequential recurrence.
__global__ __launch_bounds__(256)
void k_scan(const float* __restrict__ dt,
            const float* __restrict__ u32,
            const float* __restrict__ xd,       // 16384 x 128 (cols 64..95 = Bm|Cm)
            const float* __restrict__ xz,       // z at col 2048+d, stride 4096
            const float* __restrict__ A_log,
            const float* __restrict__ Dp,
            unsigned short* __restrict__ ybf) {
    const int tid = threadIdx.x;
    const int b   = blockIdx.y;
    const int d   = blockIdx.x * 256 + tid;

    float A[16], h[16];
#pragma unroll
    for (int s = 0; s < 16; ++s) {
        A[s] = -__expf(A_log[(size_t)d * 16 + s]);
        h[s] = 0.f;
    }
    const float dp = Dp[d];

    __shared__ __align__(16) float sBC[2][64][32];   // 64 steps x (Bm[16]|Cm[16])

    // 512 x 16B chunks per 64-step window, 2 per thread
    auto issue_chunk = [&](int l0, int buf) {
#pragma unroll
        for (int i = 0; i < 2; ++i) {
            int e = i * 256 + tid;
            int step = e >> 3, part = (e & 7) * 4;
            async_b128(&sBC[buf][step][part],
                       xd + ((size_t)b * 2048 + l0 + step) * 128 + 64 + part);
        }
    };

    auto compute = [&](int l0, int buf) {
        for (int s = 0; s < 64; ++s) {
            size_t row = (size_t)b * 2048 + l0 + s;
            float dtv = dt[row * 2048 + d];
            float uv  = u32[row * 2048 + d];
            float du  = dtv * uv;
            float y = 0.f;
#pragma unroll
            for (int j = 0; j < 16; ++j) {
                float hn = h[j] * __expf(dtv * A[j]) + du * sBC[buf][s][j];
                h[j] = hn;
                y += hn * sBC[buf][s][16 + j];
            }
            float zv = xz[row * 4096 + 2048 + d];
            float yg = (y + uv * dp) * silu(zv);
            ybf[row * 2048 + d] = f2bf(yg);
        }
    };

    issue_chunk(0, 0);
    int buf = 0, l0 = 0;
    for (; l0 + 64 < 2048; l0 += 64, buf ^= 1) {   // branch-free steady state
        issue_chunk(l0 + 64, buf ^ 1);
        wait_asynccnt<2>();
        __syncthreads();
        compute(l0, buf);
        __syncthreads();
    }
    wait_asynccnt<0>();                            // peeled last chunk
    __syncthreads();
    compute(l0, buf);
}

// ---------------- launch ------------------------------------------------------
extern "C" void kernel_launch(void* const* d_in, const int* in_sizes, int n_in,
                              void* d_out, int out_size, void* d_ws, size_t ws_size,
                              hipStream_t stream) {
    const float* x      = (const float*)d_in[0];
    const float* ln_w   = (const float*)d_in[1];
    const float* ln_b   = (const float*)d_in[2];
    const float* W_in   = (const float*)d_in[3];
    const float* conv_w = (const float*)d_in[4];
    const float* conv_b = (const float*)d_in[5];
    const float* W_x    = (const float*)d_in[6];
    const float* W_dt   = (const float*)d_in[7];
    const float* b_dt   = (const float*)d_in[8];
    const float* A_log  = (const float*)d_in[9];
    const float* Dp     = (const float*)d_in[10];
    const float* W_out  = (const float*)d_in[11];
    float* out = (float*)d_out;

    const size_t BL = 16384;                 // B*L
    char* ws = (char*)d_ws;
    size_t off = 0;
    auto alloc = [&](size_t bytes) -> char* {
        char* p = ws + off;
        off += (bytes + 255) & ~(size_t)255;
        return p;
    };
    unsigned short* xn_bf  = (unsigned short*)alloc(BL * 1024 * 2);
    unsigned short* winT   = (unsigned short*)alloc((size_t)4096 * 1024 * 2);
    unsigned short* wxT    = (unsigned short*)alloc((size_t)128 * 2048 * 2);
    unsigned short* wdtT   = (unsigned short*)alloc((size_t)2048 * 64 * 2);
    unsigned short* woutT  = (unsigned short*)alloc((size_t)1024 * 2048 * 2);
    float*          xzbuf  = (float*)alloc(BL * 4096 * 4);
    float*          u32buf = (float*)alloc(BL * 2048 * 4);
    unsigned short* ubf    = (unsigned short*)alloc(BL * 2048 * 2);
    float*          xdbuf  = (float*)alloc(BL * 128 * 4);
    unsigned short* xd_bf  = (unsigned short*)alloc(BL * 128 * 2);
    float*          dtbuf  = (float*)alloc(BL * 2048 * 4);
    unsigned short* ybf    = (unsigned short*)alloc(BL * 2048 * 2);

    auto blocks = [](size_t n) { return (unsigned)((n + 255) / 256); };

    // 1) weights -> bf16, transposed to (N x K) so GEMM B-tiles are K-contiguous
    k_f32_to_bf16_T<<<blocks((size_t)4096 * 1024), 256, 0, stream>>>(W_in,  winT, 1024, 4096, 4096);
    k_f32_to_bf16_T<<<blocks((size_t)128 * 2048),  256, 0, stream>>>(W_x,   wxT,  2048, 96,   128);
    k_f32_to_bf16_T<<<blocks((size_t)2048 * 64),   256, 0, stream>>>(W_dt,  wdtT, 64,   2048, 2048);
    k_f32_to_bf16_T<<<blocks((size_t)1024 * 2048), 256, 0, stream>>>(W_out, woutT, 2048, 1024, 1024);

    // 2) layernorm -> bf16 xn
    k_layernorm_bf16<<<(unsigned)BL, 256, 0, stream>>>(x, ln_w, ln_b, xn_bf);

    // 3) xz = xn @ W_in   (16384x1024 @ 1024x4096)
    k_gemm_bf16<0><<<dim3(4096 / 128, (unsigned)(BL / 128)), 256, 0, stream>>>(
        xn_bf, winT, xzbuf, (int)BL, 4096, 1024, 1024, 4096, nullptr, nullptr);

    // 4) depthwise conv + SiLU on u
    k_conv_silu<<<blocks(BL * 2048), 256, 0, stream>>>(xzbuf, conv_w, conv_b, u32buf, ubf);

    // 5) xd = u @ W_x_padded  (16384x2048 @ 2048x128)
    k_gemm_bf16<0><<<dim3(1, (unsigned)(BL / 128)), 256, 0, stream>>>(
        ubf, wxT, xdbuf, (int)BL, 128, 2048, 2048, 128, nullptr, nullptr);

    // 6) xd -> bf16 (A operand for dt GEMM, K=64 within lda=128)
    k_f32_to_bf16<<<blocks(BL * 128), 256, 0, stream>>>(xdbuf, xd_bf, BL * 128);

    // 7) dt = softplus(xd[:, :64] @ W_dt + b_dt)   (16384x64 @ 64x2048)
    k_gemm_bf16<1><<<dim3(2048 / 128, (unsigned)(BL / 128)), 256, 0, stream>>>(
        xd_bf, wdtT, dtbuf, (int)BL, 2048, 64, 128, 2048, b_dt, nullptr);

    // 8) selective scan fused with +u*Dp and *silu(z); emits bf16 y
    k_scan<<<dim3(2048 / 256, 8), 256, 0, stream>>>(
        dtbuf, u32buf, xdbuf, xzbuf, A_log, Dp, ybf);

    // 9) out = y @ W_out + x    (16384x2048 @ 2048x1024)
    k_gemm_bf16<2><<<dim3(1024 / 128, (unsigned)(BL / 128)), 256, 0, stream>>>(
        ybf, woutT, out, (int)BL, 1024, 2048, 2048, 1024, nullptr, x);
}
